// EigenAttention_70712341562129
// MI455X (gfx1250) — compile-verified, matrix operand order checked by reading, not run on hardware
//
#include <hip/hip_runtime.h>
#include <hip/hip_bf16.h>

typedef __attribute__((ext_vector_type(16))) _Float16 v16h;
typedef __attribute__((ext_vector_type(8)))  _Float16 v8h;
typedef __attribute__((ext_vector_type(8)))  float    v8f;
typedef __attribute__((ext_vector_type(4)))  int      v4i;

typedef __attribute__((address_space(1))) v4i* gp_v4i;  // global v4i*
typedef __attribute__((address_space(3))) v4i* lp_v4i;  // LDS v4i*

static constexpr int BB    = 4;
static constexpr int LL    = 1024;
static constexpr int DIMM  = 1024;
static constexpr int NHEAD = 16;
static constexpr int HD    = 64;
static constexpr int NTHR  = 256;

// ---------------------------------------------------------------------------
// WMMA helper: D = A(16x32 f16) * B(32x16 f16) + C(16x16 f32)
// ---------------------------------------------------------------------------
__device__ __forceinline__ v8f wmma_f16(v16h a, v16h b, v8f c) {
  return __builtin_amdgcn_wmma_f32_16x16x32_f16(
      /*neg_a=*/false, a, /*neg_b=*/false, b,
      /*c_mod=*/(short)0, c, /*reuse_a=*/false, /*reuse_b=*/false);
}

// Fragment loader (global or LDS after addrspace inference) for 16-bit
// A (16 rows x 32 K) / B (32 K x 16 cols, one contiguous K-row per column).
//   lanes 0-15 : row/col = lane,    K = {0..7, 16..23}
//   lanes 16-31: row/col = lane-16, K = {8..15, 24..31}
__device__ __forceinline__ v16h load_frag16(const _Float16* p, int ld,
                                            int lane) {
  const int row  = lane & 15;
  const int koff = (lane >> 4) << 3;
  const _Float16* r = p + (size_t)row * ld + koff;
  v8h lo = *(const v8h*)(r);
  v8h hi = *(const v8h*)(r + 16);
  v16h out;
#pragma unroll
  for (int t = 0; t < 8; ++t) { out[t] = lo[t]; out[t + 8] = hi[t]; }
  return out;
}

// ---------------------------------------------------------------------------
// Async Global -> LDS copy (16B per lane), tracked by ASYNCcnt.
// Builtin signature (verified by compiler): (v4i AS1*, v4i AS3*, Ii, Ii).
// ---------------------------------------------------------------------------
__device__ __forceinline__ void cp_g2l_b128(_Float16* ldst,
                                            const _Float16* gsrc) {
#if __has_builtin(__builtin_amdgcn_global_load_async_to_lds_b128)
  __builtin_amdgcn_global_load_async_to_lds_b128(
      (gp_v4i)(gsrc), (lp_v4i)(ldst), 0, 0);
#else
  *(v8h*)ldst = *(const v8h*)gsrc;
#endif
}

__device__ __forceinline__ void async_wait0() {
#if __has_builtin(__builtin_amdgcn_s_wait_asynccnt)
  __builtin_amdgcn_s_wait_asynccnt(0);
#else
  asm volatile("s_wait_asynccnt 0x0" ::: "memory");
#endif
}

// Streaming (write-once) output store: non-temporal hint keeps the 256 MB
// attention stream from displacing K/V/W tiles in L2.
__device__ __forceinline__ void store_nt_f32(float* p, float v) {
#if __has_builtin(__builtin_nontemporal_store)
  __builtin_nontemporal_store(v, p);
#else
  *p = v;
#endif
}

__device__ __forceinline__ float logit_xform(float t) {
  // |sim| < 1e-3 -> 0 ; *4 ; negatives clamped to >= -10
  t = (fabsf(t) < 1e-3f) ? 0.0f : t;
  t *= 4.0f;
  if (t < 0.0f) t = fmaxf(t, -10.0f);
  return t;
}

// ---------------------------------------------------------------------------
// f32 -> f16 conversion
// ---------------------------------------------------------------------------
__global__ void cvt_f16_kernel(const float* __restrict__ src,
                               _Float16* __restrict__ dst, int n) {
  int i = blockIdx.x * blockDim.x + threadIdx.x;
  if (i < n) dst[i] = (_Float16)src[i];
}

// ---------------------------------------------------------------------------
// Projection GEMM:  y[m,n] = sum_k X[m,k] * W[n,k] + bias[n]
// mode 0: store Q*metric  -> [bh][l][d]   (f16)
// mode 1: store K         -> [bh][l][d]   (f16)
// mode 2: store V^T       -> [bh][d][l]   (f16)
// mode 3: store f32       -> [m][n]       (final output projection, NT)
// ---------------------------------------------------------------------------
__global__ __launch_bounds__(NTHR)
void proj_wmma_kernel(const _Float16* __restrict__ X,
                      const _Float16* __restrict__ W,
                      const float* __restrict__ bias, int mode,
                      _Float16* __restrict__ dsth, float* __restrict__ dstf) {
  const int lane = threadIdx.x & 31;
  const int wave = threadIdx.x >> 5;
  const int m0 = blockIdx.x * 128 + wave * 16;
  const int n0 = blockIdx.y * 64;

  v8f acc[4];
#pragma unroll
  for (int n = 0; n < 4; ++n)
#pragma unroll
    for (int r = 0; r < 8; ++r) acc[n][r] = 0.0f;

  const _Float16* xrow = X + (size_t)m0 * DIMM;
#pragma unroll 2
  for (int k0 = 0; k0 < DIMM; k0 += 32) {
    __builtin_prefetch(xrow + k0 + 128, 0, 1);
    v16h a = load_frag16(xrow + k0, DIMM, lane);
#pragma unroll
    for (int n = 0; n < 4; ++n) {
      v16h bf = load_frag16(W + (size_t)(n0 + n * 16) * DIMM + k0, DIMM, lane);
      acc[n] = wmma_f16(a, bf, acc[n]);
    }
  }

  const int Mofs = (lane >> 4) << 3;
  const int N = lane & 15;
#pragma unroll
  for (int n = 0; n < 4; ++n) {
    const int col = n0 + n * 16 + N;
    const float bv = bias[col];
    const int hh = col >> 6;
    const int d  = col & 63;
#pragma unroll
    for (int r = 0; r < 8; ++r) {
      const int row  = m0 + Mofs + r;
      const int bidx = row >> 10;
      const int l    = row & 1023;
      float y = acc[n][r] + bv;
      if (mode == 0) {
        float qv = (d == 0) ? y : -y;  // Minkowski metric pre-applied to Q
        dsth[((size_t)(bidx * NHEAD + hh) * LL + l) * HD + d] = (_Float16)qv;
      } else if (mode == 1) {
        dsth[((size_t)(bidx * NHEAD + hh) * LL + l) * HD + d] = (_Float16)y;
      } else if (mode == 2) {
        dsth[((size_t)(bidx * NHEAD + hh) * HD + d) * LL + l] = (_Float16)y;
      } else {
        store_nt_f32(&dstf[(size_t)row * DIMM + col], y);
      }
    }
  }
}

// ---------------------------------------------------------------------------
// Fused attention. Block = 8 waves = 128 query rows of one (b,h).
// K/V tiles are staged ONCE per block into LDS with async global->LDS copies
// (ASYNCcnt) and shared by all 8 waves; pass-A K tiles are double-buffered so
// the async copy overlaps WMMA + online-softmax math.
//   pass A: QK^T logits (WMMA) -> online softmax stats (butterfly shuffles)
//   pass B: recompute tile, write normalized attn exactly once with NT stores
//           (256 MB write floor), LDS-transpose attn C-frag -> A-frag,
//           accumulate attn @ V (WMMA).
// ---------------------------------------------------------------------------
__global__ __launch_bounds__(NTHR)
void attn_softmax_kernel(const _Float16* __restrict__ qm,
                         const _Float16* __restrict__ kh,
                         const _Float16* __restrict__ vt,
                         float* __restrict__ attn,
                         _Float16* __restrict__ yh) {
  const int tid  = threadIdx.x;
  const int lane = tid & 31;
  const int wave = tid >> 5;
  const int bh = blockIdx.x;
  const int bb = bh >> 4;
  const int hh = bh & 15;
  const int i0 = blockIdx.y * 128 + wave * 16;

  __shared__ __align__(16) _Float16 kbufA[2][16 * HD];  // pass A ping-pong
  __shared__ __align__(16) _Float16 kbufB[32 * HD];     // pass B K tile
  __shared__ __align__(16) _Float16 vbuf[HD * 32];      // pass B V^T tile
  __shared__ __align__(16) _Float16 pbuf[8][16 * 34];   // per-wave attn tile

  const _Float16* qrow  = qm + ((size_t)bh * LL + i0) * HD;
  const _Float16* kbase = kh + (size_t)bh * LL * HD;
  const _Float16* vbase = vt + (size_t)bh * HD * LL;

  // Q fragments (K = 64 -> two 32-wide chunks), resident all kernel
  v16h aq0 = load_frag16(qrow, HD, lane);
  v16h aq1 = load_frag16(qrow + 32, HD, lane);

  // -------- pass A: online softmax statistics ------------------------------
  float m[8], s[8];
#pragma unroll
  for (int r = 0; r < 8; ++r) { m[r] = -3.0e38f; s[r] = 0.0f; }

  // stage K tile 0 (16x64 f16 = 128 x 16B chunks)
  if (tid < 128) cp_g2l_b128(&kbufA[0][tid * 8], kbase + tid * 8);
  async_wait0();
  __syncthreads();

  for (int jt = 0; jt < LL / 16; ++jt) {
    const int cur = jt & 1;
    if (jt + 1 < LL / 16 && tid < 128)  // overlap next tile with compute
      cp_g2l_b128(&kbufA[cur ^ 1][tid * 8],
                  kbase + (size_t)(jt + 1) * 16 * HD + tid * 8);

    v16h bk0 = load_frag16(&kbufA[cur][0], HD, lane);
    v16h bk1 = load_frag16(&kbufA[cur][32], HD, lane);
    v8f c;
#pragma unroll
    for (int r = 0; r < 8; ++r) c[r] = 0.0f;
    c = wmma_f16(aq0, bk0, c);
    c = wmma_f16(aq1, bk1, c);
#pragma unroll
    for (int r = 0; r < 8; ++r) {
      float t = logit_xform(c[r]);
      float tm = t;  // row max across 16 columns (16-lane butterfly)
#pragma unroll
      for (int o = 1; o < 16; o <<= 1) tm = fmaxf(tm, __shfl_xor(tm, o, 32));
      float nm = fmaxf(m[r], tm);
      float e = __expf(t - nm);
#pragma unroll
      for (int o = 1; o < 16; o <<= 1) e += __shfl_xor(e, o, 32);
      s[r] = s[r] * __expf(m[r] - nm) + e;
      m[r] = nm;
    }
    async_wait0();   // next tile resident
    __syncthreads(); // all waves done with kbufA[cur]
  }
  float rs[8];
#pragma unroll
  for (int r = 0; r < 8; ++r) rs[r] = 1.0f / s[r];

  // -------- pass B: write attn once + accumulate attn @ V ------------------
  float* arow = attn + ((size_t)bh * LL + i0) * LL;
  const int Mofs = (lane >> 4) << 3;
  const int N = lane & 15;

  v8f acc[4];
#pragma unroll
  for (int n = 0; n < 4; ++n)
#pragma unroll
    for (int r = 0; r < 8; ++r) acc[n][r] = 0.0f;

  for (int j0 = 0; j0 < LL; j0 += 32) {
    // stage K (32x64: 256 chunks) + V^T (64x32: 256 chunks), 2 chunks/thread
    cp_g2l_b128(&kbufB[tid * 8], kbase + (size_t)j0 * HD + tid * 8);
    {
      const int d = tid >> 2, ch = tid & 3;
      cp_g2l_b128(&vbuf[d * 32 + ch * 8],
                  vbase + (size_t)d * LL + j0 + ch * 8);
    }
    async_wait0();
    __syncthreads();

#pragma unroll
    for (int half = 0; half < 2; ++half) {
      const int jj = j0 + half * 16;
      v16h bk0 = load_frag16(&kbufB[half * 16 * HD], HD, lane);
      v16h bk1 = load_frag16(&kbufB[half * 16 * HD + 32], HD, lane);
      v8f c;
#pragma unroll
      for (int r = 0; r < 8; ++r) c[r] = 0.0f;
      c = wmma_f16(aq0, bk0, c);
      c = wmma_f16(aq1, bk1, c);
#pragma unroll
      for (int r = 0; r < 8; ++r) {
        float t = logit_xform(c[r]);
        float p = __expf(t - m[r]) * rs[r];
        store_nt_f32(&arow[(size_t)(Mofs + r) * LL + jj + N], p);  // attn
        pbuf[wave][(Mofs + r) * 34 + half * 16 + N] = (_Float16)p;
      }
    }
    __syncthreads();  // pbuf visible (per-wave region, cross-lane)

    // C-fragment -> A-fragment transpose through LDS
    v16h ap;
    {
      const int row  = lane & 15;
      const int koff = (lane >> 4) << 3;
#pragma unroll
      for (int t = 0; t < 8; ++t) {
        ap[t]     = pbuf[wave][row * 34 + koff + t];
        ap[t + 8] = pbuf[wave][row * 34 + koff + 16 + t];
      }
    }
#pragma unroll
    for (int n = 0; n < 4; ++n) {
      v16h bv = load_frag16(&vbuf[n * 16 * 32], 32, lane);
      acc[n] = wmma_f16(ap, bv, acc[n]);
    }
    __syncthreads();  // all reads done before next iteration's staging
  }

  // epilogue: heads re-interleaved -> yh[(b*L + l)][h*64 + d]  (f16)
#pragma unroll
  for (int n = 0; n < 4; ++n)
#pragma unroll
    for (int r = 0; r < 8; ++r) {
      const int row = i0 + Mofs + r;
      const int col = hh * 64 + n * 16 + N;
      yh[(size_t)(bb * LL + row) * DIMM + col] = (_Float16)acc[n][r];
    }
}

// ---------------------------------------------------------------------------
// Host launcher
// ---------------------------------------------------------------------------
extern "C" void kernel_launch(void* const* d_in, const int* in_sizes, int n_in,
                              void* d_out, int out_size, void* d_ws,
                              size_t ws_size, hipStream_t stream) {
  (void)in_sizes; (void)n_in; (void)out_size; (void)ws_size;

  const float* x  = (const float*)d_in[0];
  const float* Wq = (const float*)d_in[1];
  const float* bq = (const float*)d_in[2];
  const float* Wk = (const float*)d_in[3];
  const float* bk = (const float*)d_in[4];
  const float* Wv = (const float*)d_in[5];
  const float* bv = (const float*)d_in[6];
  const float* Wo = (const float*)d_in[7];
  const float* bo = (const float*)d_in[8];

  float* outp  = (float*)d_out;                  // (B,L,DIM) f32
  float* attnp = outp + (size_t)BB * LL * DIMM;  // (B,H,L,L) f32

  const size_t MTOK = (size_t)BB * LL;               // 4096 token rows
  const size_t WSZ  = (size_t)DIMM * DIMM;           // 1M
  const size_t HSZ  = (size_t)BB * NHEAD * LL * HD;  // 4M

  _Float16* xh  = (_Float16*)d_ws;   // x in f16            (4M)
  _Float16* wqh = xh  + MTOK * DIMM; // weights f16         (1M each)
  _Float16* wkh = wqh + WSZ;
  _Float16* wvh = wkh + WSZ;
  _Float16* woh = wvh + WSZ;
  _Float16* qmh = woh + WSZ;         // Q*metric [bh][l][d] (4M)
  _Float16* khh = qmh + HSZ;         // K        [bh][l][d] (4M)
  _Float16* vth = khh + HSZ;         // V^T      [bh][d][l] (4M)
  _Float16* yhh = vth + HSZ;         // attn@V   [m][n]     (4M)

  {
    int n = (int)(MTOK * DIMM);
    cvt_f16_kernel<<<(n + NTHR - 1) / NTHR, NTHR, 0, stream>>>(x, xh, n);
    n = (int)WSZ;
    cvt_f16_kernel<<<(n + NTHR - 1) / NTHR, NTHR, 0, stream>>>(Wq, wqh, n);
    cvt_f16_kernel<<<(n + NTHR - 1) / NTHR, NTHR, 0, stream>>>(Wk, wkh, n);
    cvt_f16_kernel<<<(n + NTHR - 1) / NTHR, NTHR, 0, stream>>>(Wv, wvh, n);
    cvt_f16_kernel<<<(n + NTHR - 1) / NTHR, NTHR, 0, stream>>>(Wo, woh, n);
  }

  dim3 gproj(32, 16);  // M/128 x N/64
  dim3 blk(NTHR);
  proj_wmma_kernel<<<gproj, blk, 0, stream>>>(xh, wqh, bq, 0, qmh, nullptr);
  proj_wmma_kernel<<<gproj, blk, 0, stream>>>(xh, wkh, bk, 1, khh, nullptr);
  proj_wmma_kernel<<<gproj, blk, 0, stream>>>(xh, wvh, bv, 2, vth, nullptr);

  attn_softmax_kernel<<<dim3(BB * NHEAD, LL / 128), blk, 0, stream>>>(
      qmh, khh, vth, attnp, yhh);

  proj_wmma_kernel<<<gproj, blk, 0, stream>>>(yhh, woh, bo, 3, nullptr, outp);
}
